// NMTDecoder_16947940950368
// MI455X (gfx1250) — matrix operand order repaired and checked by reading, not compiled, not run on hardware
//
#include <hip/hip_runtime.h>
#include <hip/hip_bf16.h>

// Problem constants (match reference)
#define BB 64
#define SS 64
#define TT 64
#define VV 32000
#define EE 256
#define HH 512
#define H3 1536
#define H2 1024

typedef __bf16 bf16;
typedef __attribute__((ext_vector_type(16))) __bf16 v16bf;
typedef __attribute__((ext_vector_type(8)))  float  v8f;

// ---------------------------------------------------------------------------
// Generic bf16 WMMA GEMM:  C[M,N] = A[M,K] * Wsel[N,K]^T + biasSel
// A, W row-major bf16 (K contiguous). C row-major fp32.
// Wave tile = (MT*16) x 64.  Block = 4 waves -> (MT*16) x 256.
// Dual-weight mode: rows >= mSplit use W2/bias2 (for fused [ctx;h] gate GEMM).
// mSplit must be a multiple of MT*16 so a block never straddles the split.
// NT: nontemporal C stores (write-once giant outputs; keeps W resident in L2).
// Requires: M%(MT*16)==0, N%256==0, K%32==0, buffers 16B-aligned.
// ---------------------------------------------------------------------------
template <int MT, bool NT>
__global__ __launch_bounds__(128)
void gemm_bf16_wmma(const bf16* __restrict__ A,
                    const bf16* __restrict__ W, const bf16* __restrict__ W2,
                    const float* __restrict__ bias, const float* __restrict__ bias2,
                    float* __restrict__ C, int M, int N, int K, int mSplit) {
  (void)M;
  const int lane  = threadIdx.x & 31;
  const int wave  = threadIdx.x >> 5;
  const int nGrps = N >> 8;
  const int mBlk  = blockIdx.x / nGrps;
  const int nBase = ((blockIdx.x % nGrps) << 8) + (wave << 6);
  const int lm = lane & 15;
  const int kh = lane >> 4;                  // K-half selector
  const int rowBase0 = mBlk * (MT * 16);

  const bf16*  Wu = (rowBase0 < mSplit) ? W : W2;        // uniform per block
  const float* bu = (rowBase0 < mSplit) ? bias : bias2;

  v8f acc[MT][4] = {};

  const bf16* aRow[MT];
#pragma unroll
  for (int mt = 0; mt < MT; ++mt)
    aRow[mt] = A + (size_t)(rowBase0 + mt * 16 + lm) * K + 16 * kh;

  const bf16* w0 = Wu + (size_t)(nBase + 0  + lm) * K + 16 * kh;
  const bf16* w1 = Wu + (size_t)(nBase + 16 + lm) * K + 16 * kh;
  const bf16* w2 = Wu + (size_t)(nBase + 32 + lm) * K + 16 * kh;
  const bf16* w3 = Wu + (size_t)(nBase + 48 + lm) * K + 16 * kh;

  union Frag { v16bf v; uint4 q[2]; };

  for (int kb = 0; kb < K; kb += 32) {
    // Stream-prefetch the W operand two chunks (64 k) ahead, WGP scope (locality=3).
    if (kb + 64 < K) {
      __builtin_prefetch(w0 + kb + 64, 0, 3);
      __builtin_prefetch(w1 + kb + 64, 0, 3);
      __builtin_prefetch(w2 + kb + 64, 0, 3);
      __builtin_prefetch(w3 + kb + 64, 0, 3);
    }
    // Hoist ALL fragment loads (B then A) so they issue as one clause and
    // the 4*MT WMMAs below drain them with staggered partial waits.
    Frag b0, b1, b2, b3;
    b0.q[0] = *(const uint4*)(w0 + kb);  b0.q[1] = *(const uint4*)(w0 + kb + 8);
    b1.q[0] = *(const uint4*)(w1 + kb);  b1.q[1] = *(const uint4*)(w1 + kb + 8);
    b2.q[0] = *(const uint4*)(w2 + kb);  b2.q[1] = *(const uint4*)(w2 + kb + 8);
    b3.q[0] = *(const uint4*)(w3 + kb);  b3.q[1] = *(const uint4*)(w3 + kb + 8);
    Frag a[MT];
#pragma unroll
    for (int mt = 0; mt < MT; ++mt) {
      a[mt].q[0] = *(const uint4*)(aRow[mt] + kb);
      a[mt].q[1] = *(const uint4*)(aRow[mt] + kb + 8);
    }
#pragma unroll
    for (int mt = 0; mt < MT; ++mt) {
      acc[mt][0] = __builtin_amdgcn_wmma_f32_16x16x32_bf16(false, a[mt].v, false, b0.v, (short)0, acc[mt][0], false, false);
      acc[mt][1] = __builtin_amdgcn_wmma_f32_16x16x32_bf16(false, a[mt].v, false, b1.v, (short)0, acc[mt][1], false, false);
      acc[mt][2] = __builtin_amdgcn_wmma_f32_16x16x32_bf16(false, a[mt].v, false, b2.v, (short)0, acc[mt][2], false, false);
      acc[mt][3] = __builtin_amdgcn_wmma_f32_16x16x32_bf16(false, a[mt].v, false, b3.v, (short)0, acc[mt][3], false, false);
    }
  }

#pragma unroll
  for (int mt = 0; mt < MT; ++mt) {
    const int rowBase = rowBase0 + mt * 16 + 8 * kh;
#pragma unroll
    for (int j = 0; j < 4; ++j) {
      const int nCol = nBase + 16 * j + lm;
      const float bv = bu ? bu[nCol] : 0.0f;
      float* cr = C + (size_t)rowBase * N + nCol;
#pragma unroll
      for (int i = 0; i < 8; ++i) {
        float v = acc[mt][j][i] + bv;
        if (NT) __builtin_nontemporal_store(v, cr + (size_t)i * N);
        else    cr[(size_t)i * N] = v;
      }
    }
  }
}

// ---------------------------------------------------------------------------
// fp32 -> bf16 column-slice convert:  dst[r,c] = src[r, colOff + c]
// ---------------------------------------------------------------------------
__global__ void convert_slice_bf16(const float* __restrict__ src, int srcLd, int colOff,
                                   bf16* __restrict__ dst, int rows, int cols) {
  long long i = (long long)blockIdx.x * blockDim.x + threadIdx.x;
  if (i >= (long long)rows * cols) return;
  int r = (int)(i / cols), c = (int)(i % cols);
  dst[i] = (bf16)src[(size_t)r * srcLd + colOff + c];
}

// X[t*B+b, e] = emb[y[b,t], e]  (row 0 forced to zero -> padding_idx)
__global__ void gather_embed(const float* __restrict__ emb, const int* __restrict__ tgt,
                             bf16* __restrict__ X) {
  long long i = (long long)blockIdx.x * blockDim.x + threadIdx.x;
  if (i >= (long long)TT * BB * EE) return;
  int e = (int)(i % EE);
  int row = (int)(i / EE);
  int t = row / BB, b = row % BB;
  int y = tgt[b * TT + t];
  float v = (y == 0) ? 0.0f : emb[(size_t)y * EE + e];
  X[i] = (bf16)v;
}

// h0 = initial_hidden @ W_map^T + b_map ; writes fp32 state + bf16 state row.
// state_bf layout: rows 0..63 = ctx (bf16), rows 64..127 = h (bf16).
__global__ void hidden_map(const float* __restrict__ init, const float* __restrict__ Wm,
                           const float* __restrict__ bm, float* __restrict__ h,
                           bf16* __restrict__ state_bf) {
  int i = blockIdx.x * blockDim.x + threadIdx.x;
  if (i >= BB * HH) return;
  int b = i / HH, n = i % HH;
  const float* x = init + (size_t)b * HH;
  const float* w = Wm + (size_t)n * HH;
  float s = bm[n];
  for (int k = 0; k < HH; ++k) s += x[k] * w[k];
  h[i] = s;
  state_bf[(size_t)(BB + b) * HH + n] = (bf16)s;
}

__global__ void zero_ctx(bf16* __restrict__ state_bf) {
  int i = blockIdx.x * blockDim.x + threadIdx.x;
  if (i >= BB * HH) return;
  state_bf[i] = (bf16)0.0f;   // ctx rows
}

// ---------------------------------------------------------------------------
// Fused GRU gates + attention for one timestep. One block per batch element.
// G = [ctx@W_ihc^T ; h@W_hh^T + b_hh]  (128 x 1536, from the dual GEMM).
// h_new is kept in LDS between the two phases.
// ---------------------------------------------------------------------------
__global__ __launch_bounds__(256)
void gru_attn_step(const float* __restrict__ gi_x, const float* __restrict__ G,
                   float* __restrict__ h_f, bf16* __restrict__ state_bf,
                   const float* __restrict__ enc, bf16* __restrict__ pred, int t) {
  __shared__ float hs[HH];
  __shared__ float part[4][SS];
  __shared__ float p[SS];
  __shared__ float ssum;
  const int b = blockIdx.x;
  const int tid = threadIdx.x;

  // Phase 1: gates
  const float* gx = gi_x + (size_t)(t * BB + b) * H3;   // includes b_ih
  const float* g1 = G + (size_t)b * H3;                 // ctx @ W_ihc^T
  const float* g2 = G + (size_t)(BB + b) * H3;          // h @ W_hh^T + b_hh
  for (int j = tid; j < HH; j += 256) {
    float ir = gx[j]          + g1[j];
    float iz = gx[HH + j]     + g1[HH + j];
    float in = gx[2 * HH + j] + g1[2 * HH + j];
    float r = 1.0f / (1.0f + __expf(-(ir + g2[j])));
    float z = 1.0f / (1.0f + __expf(-(iz + g2[HH + j])));
    float n = tanhf(in + r * g2[2 * HH + j]);
    float hnew = (1.0f - z) * n + z * h_f[(size_t)b * HH + j];
    h_f[(size_t)b * HH + j] = hnew;
    hs[j] = hnew;
    state_bf[(size_t)(BB + b) * HH + j] = (bf16)hnew;
    pred[(size_t)(b * TT + t) * H2 + HH + j] = (bf16)hnew;  // h-half of pred
  }
  __syncthreads();

  // Phase 2: attention (scores -> softmax -> context)
  const float* eb = enc + (size_t)b * SS * HH;
  const int s = tid & 63, seg = tid >> 6;     // 4 K-segments of 128
  float acc = 0.0f;
  const float* er = eb + (size_t)s * HH + seg * 128;
  for (int k = 0; k < 128; ++k) acc += er[k] * hs[seg * 128 + k];
  part[seg][s] = acc;
  __syncthreads();

  if (tid < SS) p[tid] = part[0][tid] + part[1][tid] + part[2][tid] + part[3][tid];
  __syncthreads();
  if (tid == 0) {
    float m = p[0];
    for (int k = 1; k < SS; ++k) m = fmaxf(m, p[k]);
    float sum = 0.0f;
    for (int k = 0; k < SS; ++k) { float e = __expf(p[k] - m); p[k] = e; sum += e; }
    ssum = sum;
  }
  __syncthreads();
  const float inv = 1.0f / ssum;
  for (int j = tid; j < HH; j += 256) {
    float c = 0.0f;
    for (int s2 = 0; s2 < SS; ++s2) c += p[s2] * eb[(size_t)s2 * HH + j];
    c *= inv;
    state_bf[(size_t)b * HH + j] = (bf16)c;                 // ctx rows
    pred[(size_t)(b * TT + t) * H2 + j] = (bf16)c;          // ctx-half of pred
  }
}

// ---------------------------------------------------------------------------
extern "C" void kernel_launch(void* const* d_in, const int* in_sizes, int n_in,
                              void* d_out, int out_size, void* d_ws, size_t ws_size,
                              hipStream_t stream) {
  (void)in_sizes; (void)n_in; (void)out_size; (void)ws_size;
  const float* enc   = (const float*)d_in[0];
  const float* h0in  = (const float*)d_in[1];
  const int*   tgt   = (const int*)  d_in[2];
  const float* emb   = (const float*)d_in[3];
  const float* W_ih  = (const float*)d_in[4];
  const float* b_ih  = (const float*)d_in[5];
  const float* W_hh  = (const float*)d_in[6];
  const float* b_hh  = (const float*)d_in[7];
  const float* W_map = (const float*)d_in[8];
  const float* b_map = (const float*)d_in[9];
  const float* W_cls = (const float*)d_in[10];
  const float* b_cls = (const float*)d_in[11];
  float* out = (float*)d_out;

  // workspace carve-up (256B aligned)
  char* ws = (char*)d_ws;
  size_t off = 0;
  auto take = [&](size_t bytes) -> char* {
    char* ptr = ws + off;
    off = (off + bytes + 255) & ~(size_t)255;
    return ptr;
  };
  bf16*  wcls_bf  = (bf16*)take((size_t)VV * H2 * 2);        // 65.5 MB
  bf16*  wihx_bf  = (bf16*)take((size_t)H3 * EE * 2);
  bf16*  wihc_bf  = (bf16*)take((size_t)H3 * HH * 2);
  bf16*  whh_bf   = (bf16*)take((size_t)H3 * HH * 2);
  bf16*  X_bf     = (bf16*)take((size_t)TT * BB * EE * 2);
  float* gi_x     = (float*)take((size_t)TT * BB * H3 * 4);  // 25 MB
  float* G        = (float*)take((size_t)2 * BB * H3 * 4);   // [128,1536]
  float* h_f      = (float*)take((size_t)BB * HH * 4);
  bf16*  state_bf = (bf16*)take((size_t)2 * BB * HH * 2);    // [ctx ; h]
  bf16*  pred_bf  = (bf16*)take((size_t)TT * BB * H2 * 2);   // 8.4 MB

  auto cdiv = [](long long a, long long b) { return (unsigned)((a + b - 1) / b); };

  // 1) weight conversions to bf16
  convert_slice_bf16<<<cdiv((long long)VV * H2, 256), 256, 0, stream>>>(W_cls, H2, 0, wcls_bf, VV, H2);
  convert_slice_bf16<<<cdiv((long long)H3 * EE, 256), 256, 0, stream>>>(W_ih, EE + HH, 0, wihx_bf, H3, EE);
  convert_slice_bf16<<<cdiv((long long)H3 * HH, 256), 256, 0, stream>>>(W_ih, EE + HH, EE, wihc_bf, H3, HH);
  convert_slice_bf16<<<cdiv((long long)H3 * HH, 256), 256, 0, stream>>>(W_hh, HH, 0, whh_bf, H3, HH);

  // 2) embedding gather (padding_idx=0 zeroed), initial state
  gather_embed<<<cdiv((long long)TT * BB * EE, 256), 256, 0, stream>>>(emb, tgt, X_bf);
  hidden_map<<<cdiv(BB * HH, 256), 256, 0, stream>>>(h0in, W_map, b_map, h_f, state_bf);
  zero_ctx<<<cdiv(BB * HH, 256), 256, 0, stream>>>(state_bf);

  // 3) gi_x = X @ W_ihx^T + b_ih   [4096, 1536, K=256], 64x64 wave tiles
  gemm_bf16_wmma<4, false><<<(TT * BB / 64) * (H3 / 256), 128, 0, stream>>>(
      X_bf, wihx_bf, wihx_bf, b_ih, b_ih, gi_x, TT * BB, H3, EE, TT * BB);

  // 4) sequential recurrence: 2 launches per step
  //    dual GEMM: G[0:64] = ctx@W_ihc^T ; G[64:128] = h@W_hh^T + b_hh
  const int stepBlocks = (2 * BB / 16) * (H3 / 256);  // 8 * 6 = 48
  for (int t = 0; t < TT; ++t) {
    gemm_bf16_wmma<1, false><<<stepBlocks, 128, 0, stream>>>(
        state_bf, wihc_bf, whh_bf, nullptr, b_hh, G, 2 * BB, H3, HH, BB);
    gru_attn_step<<<BB, 256, 0, stream>>>(gi_x, G, h_f, state_bf, enc, pred_bf, t);
  }

  // 5) classifier: out[(b*T+t), v] = pred @ W_cls^T + b_cls  [4096, 32000, K=1024]
  //    64x64 wave tiles, nontemporal output stores (keep W_cls hot in L2).
  gemm_bf16_wmma<4, true><<<(TT * BB / 64) * (VV / 256), 128, 0, stream>>>(
      pred_bf, wcls_bf, wcls_bf, b_cls, b_cls, out, TT * BB, VV, H2, TT * BB);
}
